// CLIPtraseRecalibration_3264175145508
// MI455X (gfx1250) — compile-verified, matrix operand order checked by reading, not run on hardware
//
#include <hip/hip_runtime.h>
#include <hip/hip_bf16.h>

#define B_SZ   8
#define N_TOK  2304
#define D_DIM  768
#define TEMP_INV 20.0f      // 1 / 0.05
#define STRENGTH 0.5f
#define JSTEP  128          // keys per iteration (8 waves x 16)
#define KCH    24           // 768 / 32  (K chunks for GEMM1)
#define ITERS  18           // 2304 / 128
#define DCH    96           // D dims per wave in GEMM2
#define NTL    6            // 96 / 16  (N tiles per wave in GEMM2)

typedef __attribute__((ext_vector_type(8)))  _Float16 v8h;
typedef __attribute__((ext_vector_type(16))) _Float16 v16h;
typedef __attribute__((ext_vector_type(8)))  float    v8f;

__device__ __forceinline__ v8f zero8() {
    v8f z;
#pragma unroll
    for (int i = 0; i < 8; ++i) z[i] = 0.0f;
    return z;
}

// Assemble a 16x32 f16 A/B fragment for this lane: two contiguous 16B chunks.
__device__ __forceinline__ v16h load_frag(const _Float16* p) {
    v8h lo = *(const v8h*)(p);
    v8h hi = *(const v8h*)(p + 16);
    return __builtin_shufflevector(lo, hi, 0,1,2,3,4,5,6,7,8,9,10,11,12,13,14,15);
}

// ---------------- Pass 1: L2-normalize rows, emit f16 xn (row-major + transposed) + norms ----
__global__ void clip_normalize_kernel(const float* __restrict__ x,
                                      _Float16* __restrict__ xn,
                                      _Float16* __restrict__ xnT,
                                      float* __restrict__ norms) {
    const int brow = blockIdx.x;            // 0 .. B*N-1
    const int b = brow / N_TOK;
    const int n = brow % N_TOK;
    const float* row = x + (size_t)brow * D_DIM;

    __shared__ float red[256];
    float ss = 0.0f;
    for (int i = threadIdx.x; i < D_DIM; i += 256) {
        float v = row[i];
        ss += v * v;
    }
    red[threadIdx.x] = ss;
    __syncthreads();
#pragma unroll
    for (int off = 128; off > 0; off >>= 1) {
        if (threadIdx.x < off) red[threadIdx.x] += red[threadIdx.x + off];
        __syncthreads();
    }
    const float nrm = fmaxf(sqrtf(red[0]), 1e-12f);
    const float inv = 1.0f / nrm;
    if (threadIdx.x == 0) norms[brow] = nrm;

    for (int i = threadIdx.x; i < D_DIM; i += 256) {
        _Float16 h = (_Float16)(row[i] * inv);
        xn [(size_t)brow * D_DIM + i] = h;
        xnT[((size_t)b * D_DIM + i) * N_TOK + n] = h;
    }
}

// ---------------- Pass 2: flash-attention style recalibration -------------------------------
__global__ void __launch_bounds__(256)
clip_recal_kernel(const float* __restrict__ x,
                  const _Float16* __restrict__ xn,
                  const _Float16* __restrict__ xnT,
                  const float* __restrict__ norms,
                  float* __restrict__ out) {
    const int b    = blockIdx.y;
    const int row0 = blockIdx.x * 16;       // 16 query rows per block
    const int tid  = threadIdx.x;
    const int w    = tid >> 5;              // wave id 0..7
    const int lane = tid & 31;
    const int ln16 = lane & 15;
    const int hf   = lane >> 4;             // K-half select for fragments

    __shared__ __align__(16) _Float16 Qbuf[16 * D_DIM];   // 24 KB: query tile (f16)
    __shared__ __align__(16) _Float16 Pbuf[16 * JSTEP];   //  4 KB: P' tile (16 x 128)
    __shared__ float Mrun[16], Lrun[16], Scale[16];
    __shared__ float red1[8 * 16], red2[8 * 16];

    if (tid < 16) { Mrun[tid] = -3.0e38f; Lrun[tid] = 0.0f; }

    // Cooperative load of the 16-row Q tile into LDS (keeps VGPR count < 256).
    {
        const v8h* src = (const v8h*)(xn + (size_t)(b * N_TOK + row0) * D_DIM);
        v8h* dst = (v8h*)Qbuf;
#pragma unroll
        for (int i = 0; i < (16 * D_DIM) / 8 / 256; ++i)
            dst[tid + i * 256] = src[tid + i * 256];
    }

    v8f acc[NTL];
#pragma unroll
    for (int t = 0; t < NTL; ++t) acc[t] = zero8();

    const int d0 = w * DCH;                 // this wave's D slice for GEMM2
    const _Float16* qlds = &Qbuf[ln16 * D_DIM + hf * 8];
    __syncthreads();

    for (int it = 0; it < ITERS; ++it) {
        const int j = it * JSTEP;

        // ---- GEMM1: S = Q . K^T for this wave's 16-key slice ----
        // Two accumulators break the WMMA->WMMA dependency chain; fragments are
        // loaded 4 K-chunks at a time so the loads issue as clauses with MLP.
        const _Float16* kbase =
            xn + ((size_t)(b * N_TOK + j + w * 16 + ln16)) * D_DIM + hf * 8;
        v8f s0 = zero8(), s1 = zero8();
#pragma unroll
        for (int g = 0; g < KCH / 4; ++g) {
            v16h af0 = load_frag(qlds + (g * 4 + 0) * 32);
            v16h af1 = load_frag(qlds + (g * 4 + 1) * 32);
            v16h af2 = load_frag(qlds + (g * 4 + 2) * 32);
            v16h af3 = load_frag(qlds + (g * 4 + 3) * 32);
            v16h bf0 = load_frag(kbase + (g * 4 + 0) * 32);
            v16h bf1 = load_frag(kbase + (g * 4 + 1) * 32);
            v16h bf2 = load_frag(kbase + (g * 4 + 2) * 32);
            v16h bf3 = load_frag(kbase + (g * 4 + 3) * 32);
            s0 = __builtin_amdgcn_wmma_f32_16x16x32_f16(false, af0, false, bf0, (short)0, s0, false, false);
            s1 = __builtin_amdgcn_wmma_f32_16x16x32_f16(false, af1, false, bf1, (short)0, s1, false, false);
            s0 = __builtin_amdgcn_wmma_f32_16x16x32_f16(false, af2, false, bf2, (short)0, s0, false, false);
            s1 = __builtin_amdgcn_wmma_f32_16x16x32_f16(false, af3, false, bf3, (short)0, s1, false, false);
        }
        v8f s = s0 + s1;
#pragma unroll
        for (int r = 0; r < 8; ++r) s[r] *= TEMP_INV;

        // ---- per-row max across this wave's 16 columns (lanes within half) ----
        float mr[8];
#pragma unroll
        for (int r = 0; r < 8; ++r) {
            float v = s[r];
            v = fmaxf(v, __shfl_xor(v, 1, 32));
            v = fmaxf(v, __shfl_xor(v, 2, 32));
            v = fmaxf(v, __shfl_xor(v, 4, 32));
            v = fmaxf(v, __shfl_xor(v, 8, 32));
            mr[r] = v;
        }
        if (ln16 == 0) {
#pragma unroll
            for (int r = 0; r < 8; ++r) red1[w * 16 + hf * 8 + r] = mr[r];
        }
        __syncthreads();

        // ---- block-wide running max + rescale factor (one thread per row) ----
        if (tid < 16) {
            float mb = red1[tid];
#pragma unroll
            for (int ww = 1; ww < 8; ++ww) mb = fmaxf(mb, red1[ww * 16 + tid]);
            float mold = Mrun[tid];
            float mnew = fmaxf(mold, mb);
            Scale[tid] = __expf(mold - mnew);
            Mrun[tid]  = mnew;
        }
        __syncthreads();

        float mn[8], sc[8];
#pragma unroll
        for (int r = 0; r < 8; ++r) {
            mn[r] = Mrun[hf * 8 + r];
            sc[r] = Scale[hf * 8 + r];
        }

        // ---- P = exp(S - m); row-sum; fold key-norm; stash P' to LDS as f16 ----
        const float nrmv = norms[b * N_TOK + j + w * 16 + ln16];  // lane-constant key norm
        float p[8], rs[8];
#pragma unroll
        for (int r = 0; r < 8; ++r) {
            p[r] = __expf(s[r] - mn[r]);
            float v = p[r];
            v += __shfl_xor(v, 1, 32);
            v += __shfl_xor(v, 2, 32);
            v += __shfl_xor(v, 4, 32);
            v += __shfl_xor(v, 8, 32);
            rs[r] = v;
        }
        if (ln16 == 0) {
#pragma unroll
            for (int r = 0; r < 8; ++r) red2[w * 16 + hf * 8 + r] = rs[r];
        }
#pragma unroll
        for (int r = 0; r < 8; ++r) {
            Pbuf[(hf * 8 + r) * JSTEP + w * 16 + ln16] = (_Float16)(p[r] * nrmv);
        }
        __syncthreads();

        if (tid < 16) {
            float ssum = 0.0f;
#pragma unroll
            for (int ww = 0; ww < 8; ++ww) ssum += red2[ww * 16 + tid];
            Lrun[tid] = Lrun[tid] * Scale[tid] + ssum;
        }

        // ---- online rescale of accumulators ----
        v8f scv;
#pragma unroll
        for (int r = 0; r < 8; ++r) scv[r] = sc[r];
#pragma unroll
        for (int t = 0; t < NTL; ++t) acc[t] *= scv;

        // ---- GEMM2: acc += P'(16x128) . V(128 x 96) from transposed xn ----
        // All 6 B fragments for a kk are loaded up front (12 b128 loads in flight),
        // then 6 independent WMMAs retire them.
#pragma unroll
        for (int kk = 0; kk < 4; ++kk) {
            const _Float16* ap = &Pbuf[ln16 * JSTEP + kk * 32 + hf * 8];
            v16h af = load_frag(ap);
            v16h bf[NTL];
#pragma unroll
            for (int t = 0; t < NTL; ++t) {
                const _Float16* vp =
                    xnT + ((size_t)(b * D_DIM + d0 + t * 16 + ln16)) * N_TOK
                        + j + kk * 32 + hf * 8;
                bf[t] = load_frag(vp);
            }
#pragma unroll
            for (int t = 0; t < NTL; ++t) {
                acc[t] = __builtin_amdgcn_wmma_f32_16x16x32_f16(
                             false, af, false, bf[t], (short)0, acc[t], false, false);
            }
        }
        __syncthreads();   // protect Pbuf / red1 / Scale for next iteration
    }

    // ---- epilogue: out = 0.5*x + 0.5*acc/l ----
    float linv[8];
#pragma unroll
    for (int r = 0; r < 8; ++r) linv[r] = 1.0f / Lrun[hf * 8 + r];

#pragma unroll
    for (int t = 0; t < NTL; ++t) {
#pragma unroll
        for (int r = 0; r < 8; ++r) {
            const int row = hf * 8 + r;
            const int dim = d0 + t * 16 + ln16;
            const size_t idx = ((size_t)(b * N_TOK + row0 + row)) * D_DIM + dim;
            out[idx] = (1.0f - STRENGTH) * x[idx] + STRENGTH * acc[t][r] * linv[r];
        }
    }
}

extern "C" void kernel_launch(void* const* d_in, const int* in_sizes, int n_in,
                              void* d_out, int out_size, void* d_ws, size_t ws_size,
                              hipStream_t stream) {
    const float* x = (const float*)d_in[0];
    float* out = (float*)d_out;

    const size_t elems = (size_t)B_SZ * N_TOK * D_DIM;
    _Float16* xn   = (_Float16*)d_ws;
    _Float16* xnT  = xn + elems;
    float*    nrms = (float*)(xnT + elems);

    hipLaunchKernelGGL(clip_normalize_kernel,
                       dim3(B_SZ * N_TOK), dim3(256), 0, stream,
                       x, xn, xnT, nrms);
    hipLaunchKernelGGL(clip_recal_kernel,
                       dim3(N_TOK / 16, B_SZ), dim3(256), 0, stream,
                       x, xn, xnT, nrms, out);
}
